// GNN_atom_aggregate_34110630265236
// MI455X (gfx1250) — matrix operand order repaired and checked by reading, not compiled, not run on hardware
//
#include <hip/hip_runtime.h>

// ---------------------------------------------------------------------------
// GATv2-style atom aggregation, fused, for gfx1250 (MI455X).
//   Phase A: af  = atom_feature @ W_a^T + b_a        (WMMA bf16, 16x256)
//   Phase B: nf  = neighbor     @ W_n^T + b_n        (WMMA bf16, 256x256) -> LDS
//            B-fragments held in registers and reused across both M-tiles.
//   Phase C: GATv2 scores + masked softmax over K=16 + weighted sum + LayerNorm
// One workgroup (256 thr = 8 wave32) per (batch b, 16-atom tile).
// ---------------------------------------------------------------------------

typedef __attribute__((ext_vector_type(16))) __bf16 v16bf;
typedef __attribute__((ext_vector_type(8)))  float  v8f;

#define WST 264            // padded LDS row stride in bf16 elems (256 data + 8 pad)
#define SHMEM_BYTES (2u * 256u * WST * 2u + 16u * 256u * 4u + 16u * 16u * 4u * 4u)

struct alignas(16) U128 { unsigned x, y, z, w; };     // POD 16B (HIP uint4 has ctors)
struct alignas(32) Frag8u { unsigned u[8]; };         // bit_cast image of v16bf
struct alignas(32) Frag2q { U128 a, b; };             // bit_cast image of v16bf

__device__ __forceinline__ unsigned fbits(float x) {
  union { float f; unsigned u; } v; v.f = x; return v.u;
}

// Pack two fp32 -> packed bf16 pair (a -> low16, b -> high16), truncating.
// Single v_perm_b32: result bytes (LSB..MSB) = {a.b2, a.b3, b.b2, b.b3}.
__device__ __forceinline__ unsigned pkbf(float a, float b) {
#if __has_builtin(__builtin_amdgcn_perm)
  return __builtin_amdgcn_perm(fbits(b), fbits(a), 0x07060302u);
#else
  return (fbits(b) & 0xffff0000u) | (fbits(a) >> 16);
#endif
}
__device__ __forceinline__ unsigned short f2bf(float a) {
  return (unsigned short)(fbits(a) >> 16);            // truncating bf16
}

// Convert one fp32 weight matrix [256][256] -> bf16 LDS [256][WST]
__device__ __forceinline__ void stage_weight(const float* __restrict__ W,
                                             unsigned short* __restrict__ dst,
                                             int tid) {
  #pragma unroll 4
  for (int i = 0; i < 64; ++i) {
    int idx4 = tid + (i << 8);            // float4 index, coalesced
    int r    = idx4 >> 6;
    int c4   = idx4 & 63;
    float4 v = ((const float4*)W)[idx4];
    unsigned long long pk = ((unsigned long long)pkbf(v.z, v.w) << 32)
                          |  (unsigned long long)pkbf(v.x, v.y);
    *(unsigned long long*)(dst + r * WST + (c4 << 2)) = pk;          // ds_store_b64
  }
}

// A fragment (16x32 bf16) for one 16-row tile from fp32 global memory.
// Contiguous-k layout (matches load_bfrag): lanes 0-15 hold M, K runs [16h,16h+16) contiguous.
__device__ __forceinline__ v16bf load_afrag(const float* __restrict__ rowp, int h) {
  const float* p0 = rowp + 16 * h;
  const float* p1 = rowp + 16 * h + 8;
  float4 a0 = *(const float4*)(p0);
  float4 a1 = *(const float4*)(p0 + 4);
  float4 b0 = *(const float4*)(p1);
  float4 b1 = *(const float4*)(p1 + 4);
  Frag8u f;
  f.u[0] = pkbf(a0.x, a0.y); f.u[1] = pkbf(a0.z, a0.w);
  f.u[2] = pkbf(a1.x, a1.y); f.u[3] = pkbf(a1.z, a1.w);
  f.u[4] = pkbf(b0.x, b0.y); f.u[5] = pkbf(b0.z, b0.w);
  f.u[6] = pkbf(b1.x, b1.y); f.u[7] = pkbf(b1.z, b1.w);
  return __builtin_bit_cast(v16bf, f);
}

// B fragment (32x16 bf16) from bf16 LDS weights [256][WST].
// Layout: lane = column n (0-15), lanes 16-31 hold K+16; K contiguous per lane.
__device__ __forceinline__ v16bf load_bfrag(const unsigned short* __restrict__ Wbf,
                                            int nt, int ks, int lane) {
  int d = nt * 16 + (lane & 15);
  const unsigned short* base = Wbf + d * WST + ks * 32 + ((lane >> 4) << 4);
  Frag2q f;
  f.a = *(const U128*)(base);             // ds_load_b128
  f.b = *(const U128*)(base + 8);         // ds_load_b128
  return __builtin_bit_cast(v16bf, f);
}

__global__ __launch_bounds__(256, 1)
void gat_wmma_kernel(const float* __restrict__ atom_feature,  // [32,128,256]
                     const float* __restrict__ anf,           // [32,128,16,256]
                     const float* __restrict__ smask,         // [32,128,16,1]
                     const float* __restrict__ amask,         // [32,128,16,1]
                     const float* __restrict__ W_a,           // [256,256]
                     const float* __restrict__ b_a,           // [256]
                     const float* __restrict__ W_n,           // [256,256]
                     const float* __restrict__ b_n,           // [256]
                     const float* __restrict__ w_align,       // [64]
                     const float* __restrict__ b_align_p,     // [1]
                     const float* __restrict__ ln_g,          // [256]
                     const float* __restrict__ ln_b,          // [256]
                     float* __restrict__ out)                 // [32,128,256]
{
  extern __shared__ char smem[];
  unsigned short* Wbf    = (unsigned short*)(smem);                        // [256][WST]
  unsigned short* nfLds  = (unsigned short*)(smem + 256u * WST * 2u);      // [256][WST]
  float*          afLds  = (float*)(smem + 2u * 256u * WST * 2u);          // [16][256]
  float*          attwLds= (float*)(smem + 2u * 256u * WST * 2u + 16384u); // [16][16][4]

  const int tid  = threadIdx.x;
  const int lane = tid & 31;
  const int wv   = tid >> 5;           // wave 0..7
  const int h    = lane >> 4;          // half-wave
  const int ln16 = lane & 15;
  const int blk  = blockIdx.x;
  const int b    = blk >> 3;
  const int n0   = (blk & 7) << 4;     // first atom of this tile

  // ---------------- Phase A: af = atom_feature @ W_a^T + b_a ----------------
  stage_weight(W_a, Wbf, tid);
  __syncthreads();
  {
    const float* Aaf = atom_feature + (size_t)(b * 128 + n0) * 256;
    const float* rowp = Aaf + ln16 * 256;
    #pragma unroll
    for (int t2 = 0; t2 < 2; ++t2) {
      int nt = wv * 2 + t2;
      v8f acc = {};
      for (int ks = 0; ks < 8; ++ks) {
        v16bf a = load_afrag(rowp + ks * 32, h);
        v16bf w = load_bfrag(Wbf, nt, ks, lane);
        acc = __builtin_amdgcn_wmma_f32_16x16x32_bf16(false, a, false, w,
                                                      (short)0, acc, false, false);
      }
      int dcol = nt * 16 + ln16;
      float bias = b_a[dcol];
      #pragma unroll
      for (int r = 0; r < 8; ++r)
        afLds[(r + 8 * h) * 256 + dcol] = acc[r] + bias;   // row = atom
    }
  }
  __syncthreads();

  // ---------------- Phase B: nf = neighbor @ W_n^T + b_n -> LDS -------------
  stage_weight(W_n, Wbf, tid);
  __syncthreads();
  {
    const float* Anf = anf + (size_t)(b * 128 + n0) * 4096;   // [256 rows][256]
    const int mt0 = wv * 2, mt1 = wv * 2 + 1;                 // M-tile == atom
    const float* rowp0 = Anf + (mt0 * 16 + ln16) * 256;
    const float* rowp1 = Anf + (mt1 * 16 + ln16) * 256;
    v8f acc0[16] = {};
    v8f acc1[16] = {};
    for (int ks = 0; ks < 8; ++ks) {
      v16bf bfr[16];
      #pragma unroll
      for (int nt = 0; nt < 16; ++nt) bfr[nt] = load_bfrag(Wbf, nt, ks, lane);
      v16bf a0 = load_afrag(rowp0 + ks * 32, h);
      #pragma unroll
      for (int nt = 0; nt < 16; ++nt)
        acc0[nt] = __builtin_amdgcn_wmma_f32_16x16x32_bf16(false, a0, false, bfr[nt],
                                                           (short)0, acc0[nt],
                                                           false, false);
      v16bf a1 = load_afrag(rowp1 + ks * 32, h);
      #pragma unroll
      for (int nt = 0; nt < 16; ++nt)
        acc1[nt] = __builtin_amdgcn_wmma_f32_16x16x32_bf16(false, a1, false, bfr[nt],
                                                           (short)0, acc1[nt],
                                                           false, false);
    }
    #pragma unroll
    for (int nt = 0; nt < 16; ++nt) {
      int dcol = nt * 16 + ln16;
      float bias = b_n[dcol];
      #pragma unroll
      for (int r = 0; r < 8; ++r) {
        int rr = r + 8 * h;
        nfLds[(mt0 * 16 + rr) * WST + dcol] = f2bf(acc0[nt][r] + bias);
        nfLds[(mt1 * 16 + rr) * WST + dcol] = f2bf(acc1[nt][r] + bias);
      }
    }
  }
  __syncthreads();

  // ---------------- Phase C1: scores + masked softmax over K ---------------
  {
    const int a = tid >> 4, k = tid & 15;           // thread = (atom, neighbor)
    const unsigned short* nrow = nfLds + tid * WST; // row a*16+k == tid
    const float* arow = afLds + a * 256;
    const float b_al = b_align_p[0];
    float s[4] = {0.f, 0.f, 0.f, 0.f};
    for (int i = 0; i < 32; ++i) {                  // 8 d's per iter
      U128 q = *(const U128*)(nrow + i * 8);
      unsigned uu[4] = {q.x, q.y, q.z, q.w};
      float av[8];
      *(float4*)(av)     = *(const float4*)(arow + i * 8);
      *(float4*)(av + 4) = *(const float4*)(arow + i * 8 + 4);
      int jb = (i * 8) & 63;
      float accs = 0.f;
      #pragma unroll
      for (int p = 0; p < 4; ++p) {
        unsigned u = uu[p];
        float n0v = __uint_as_float(u << 16);            // even elem (bf16 -> f32)
        float n1v = __uint_as_float(u & 0xffff0000u);    // odd elem
        float x0 = av[2 * p]     + n0v;
        float x1 = av[2 * p + 1] + n1v;
        x0 = fmaxf(x0, 0.2f * x0);                       // LeakyReLU (slope 0.2)
        x1 = fmaxf(x1, 0.2f * x1);
        accs += x0 * w_align[jb + 2 * p] + x1 * w_align[jb + 2 * p + 1];
      }
      s[i >> 3] += accs;
    }
    size_t mi = (size_t)(b * 128 + n0 + a) * 16 + k;
    float sm = smask[mi], am = amask[mi];
    float attv[4];
    #pragma unroll
    for (int hh = 0; hh < 4; ++hh) {
      float sc = s[hh] + b_al + sm;
      float m = sc;
      #pragma unroll
      for (int off = 8; off >= 1; off >>= 1) m = fmaxf(m, __shfl_xor(m, off, 32));
      float p = __expf(sc - m);
      float sum = p;
      #pragma unroll
      for (int off = 8; off >= 1; off >>= 1) sum += __shfl_xor(sum, off, 32);
      attv[hh] = (p / sum) * am;
    }
    float4 t; t.x = attv[0]; t.y = attv[1]; t.z = attv[2]; t.w = attv[3];
    *(float4*)(attwLds + tid * 4) = t;
  }
  __syncthreads();

  // ---------------- Phase C2: weighted sum + LayerNorm ---------------------
  {
    const int a = tid >> 4, c = tid & 15;           // thread = (atom, 16-d chunk)
    const int hsel = c >> 2;                        // head of this d-chunk
    const float* aw = attwLds + a * 64;
    float ctx[16] = {};
    for (int k = 0; k < 16; ++k) {
      float w = aw[k * 4 + hsel];
      const unsigned short* nrow = nfLds + (a * 16 + k) * WST + c * 16;
      U128 q0 = *(const U128*)(nrow);
      U128 q1 = *(const U128*)(nrow + 8);
      unsigned uu[8] = {q0.x, q0.y, q0.z, q0.w, q1.x, q1.y, q1.z, q1.w};
      #pragma unroll
      for (int p = 0; p < 8; ++p) {
        unsigned u = uu[p];
        ctx[2 * p]     += w * __uint_as_float(u << 16);
        ctx[2 * p + 1] += w * __uint_as_float(u & 0xffff0000u);
      }
    }
    float sum = 0.f, sq = 0.f;
    #pragma unroll
    for (int e = 0; e < 16; ++e) { sum += ctx[e]; sq += ctx[e] * ctx[e]; }
    #pragma unroll
    for (int off = 8; off >= 1; off >>= 1) {
      sum += __shfl_xor(sum, off, 32);
      sq  += __shfl_xor(sq,  off, 32);
    }
    float mu  = sum * (1.f / 256.f);
    float var = sq * (1.f / 256.f) - mu * mu;
    float rs  = rsqrtf(var + 1e-5f);
    float ov[16];
    #pragma unroll
    for (int e = 0; e < 16; ++e) {
      int d = c * 16 + e;
      ov[e] = (ctx[e] - mu) * rs * ln_g[d] + ln_b[d];
    }
    float* orow = out + (size_t)(b * 128 + n0 + a) * 256 + c * 16;
    #pragma unroll
    for (int q = 0; q < 4; ++q) {
      float4 t; t.x = ov[q*4]; t.y = ov[q*4+1]; t.z = ov[q*4+2]; t.w = ov[q*4+3];
      ((float4*)orow)[q] = t;
    }
  }
}

extern "C" void kernel_launch(void* const* d_in, const int* in_sizes, int n_in,
                              void* d_out, int out_size, void* d_ws, size_t ws_size,
                              hipStream_t stream) {
  (void)in_sizes; (void)n_in; (void)out_size; (void)d_ws; (void)ws_size;
  const float* atom_feature = (const float*)d_in[0];
  const float* anf          = (const float*)d_in[1];
  const float* smask        = (const float*)d_in[2];
  const float* amask        = (const float*)d_in[3];
  const float* W_a          = (const float*)d_in[4];
  const float* b_a          = (const float*)d_in[5];
  const float* W_n          = (const float*)d_in[6];
  const float* b_n          = (const float*)d_in[7];
  const float* w_align      = (const float*)d_in[8];
  const float* b_align      = (const float*)d_in[9];
  const float* ln_g         = (const float*)d_in[10];
  const float* ln_b         = (const float*)d_in[11];
  float* out = (float*)d_out;

  (void)hipFuncSetAttribute((const void*)gat_wmma_kernel,
                            hipFuncAttributeMaxDynamicSharedMemorySize,
                            (int)SHMEM_BYTES);
  dim3 grid(256), block(256);
  hipLaunchKernelGGL(gat_wmma_kernel, grid, block, SHMEM_BYTES, stream,
                     atom_feature, anf, smask, amask, W_a, b_a, W_n, b_n,
                     w_align, b_align, ln_g, ln_b, out);
}
